// TransformerBlock_53635551593116
// MI455X (gfx1250) — compile-verified, matrix operand order checked by reading, not run on hardware
//
#include <hip/hip_runtime.h>

typedef _Float16 v16h __attribute__((ext_vector_type(16)));
typedef _Float16 v8h  __attribute__((ext_vector_type(8)));
typedef float    v8f  __attribute__((ext_vector_type(8)));
typedef int      v4i  __attribute__((ext_vector_type(4)));

#define S_LEN 2048
#define DM    1024
#define HEADS 16
#define DKH   64
#define FF    4096
#define MROWS 4096   // B * S

union V16 { v16h v; v8h h[2]; };

__device__ __forceinline__ v8f wmma16(v16h a, v16h b, v8f c) {
  // D = A(16x32 f16) * B(32x16 f16) + C(16x16 f32)
  return __builtin_amdgcn_wmma_f32_16x16x32_f16(false, a, false, b, (short)0, c,
                                                false, false);
}

// ---- gfx1250 async global->LDS copy (ASYNCcnt path), guarded ----------------
#if defined(__has_builtin)
#  if __has_builtin(__builtin_amdgcn_global_load_async_to_lds_b128) && \
      __has_builtin(__builtin_amdgcn_s_wait_asynccnt)
#    define USE_ASYNC_COPY 1
#  endif
#endif
#ifndef USE_ASYNC_COPY
#  define USE_ASYNC_COPY 0
#endif

__device__ __forceinline__ void copy16_g2l(const _Float16* g, _Float16* l) {
#if USE_ASYNC_COPY
  // builtin signature (probe-derived): (int4* src_global, int4* dst_lds,
  //                                     imm offset, imm cpol)
  __builtin_amdgcn_global_load_async_to_lds_b128((v4i*)g, (v4i*)l, 0, 0);
#else
  *(v8h*)l = *(const v8h*)g;
#endif
}

__device__ __forceinline__ void copy_fence() {
#if USE_ASYNC_COPY
  __builtin_amdgcn_s_wait_asynccnt(0);
#endif
}

// ---------------------------------------------------------------- convert
__global__ __launch_bounds__(256) void cvt_f32_f16(const float* __restrict__ s,
                                                   _Float16* __restrict__ d, int n) {
  int i = blockIdx.x * 256 + threadIdx.x;
  if (i < n) d[i] = (_Float16)s[i];
}

// ---------------------------------------------------------------- f16 transpose
// out[c][r] = in[r][c]; in is [R,C]; 32x32 tiles, fully coalesced both sides.
__global__ __launch_bounds__(256) void transpose_f16(const _Float16* __restrict__ in,
                                                     _Float16* __restrict__ out,
                                                     int R, int C) {
  __shared__ _Float16 t[32][33];
  const int bx = blockIdx.x * 32, by = blockIdx.y * 32;
  const int tx = threadIdx.x & 31, ty4 = (threadIdx.x >> 5) * 4;
  #pragma unroll
  for (int i = 0; i < 4; ++i)
    t[ty4 + i][tx] = in[(size_t)(by + ty4 + i) * C + bx + tx];
  __syncthreads();
  #pragma unroll
  for (int i = 0; i < 4; ++i)
    out[(size_t)(bx + ty4 + i) * R + by + tx] = t[tx][ty4 + i];
}

// ---------------------------------------------------------------- GEMM (NT)
// C[M,N] = A[M,K] * B[N,K]^T + bias[N]
// EPI: 0 = f16 out, 1 = f32 out, 2 = exact GELU then f16 out
template<int EPI>
__global__ __launch_bounds__(256) void gemm_nt(
    const _Float16* __restrict__ A, const _Float16* __restrict__ B,
    const float* __restrict__ bias, float* __restrict__ Cf,
    _Float16* __restrict__ Ch, int M, int N, int K)
{
  constexpr int BK = 32, LDT = 40;       // stride 40 f16: conflict-free, 16B aligned
  __shared__ _Float16 As[128 * LDT];
  __shared__ _Float16 Bs[128 * LDT];
  const int tid  = threadIdx.x;
  const int lane = tid & 31, wave = tid >> 5;
  const int wm = wave >> 2, wn = wave & 3;            // 2x4 wave grid
  const int m0 = blockIdx.y * 128, n0 = blockIdx.x * 128;
  const int lrow = lane & 15;
  const int half = (lane < 16) ? 0 : 8;

  v8f acc[4][2] = {};

  for (int k0 = 0; k0 < K; k0 += BK) {
    __syncthreads();
    #pragma unroll
    for (int t = 0; t < 2; ++t) {                     // 512 16B vectors per matrix
      int vid = tid * 2 + t;
      int row = vid >> 2, c8 = (vid & 3) * 8;
      copy16_g2l(&A[(size_t)(m0 + row) * K + k0 + c8], &As[row * LDT + c8]);
      copy16_g2l(&B[(size_t)(n0 + row) * K + k0 + c8], &Bs[row * LDT + c8]);
    }
    if (k0 + BK < K) {                                // gfx1250 global_prefetch_b8
      int row = tid >> 1;
      __builtin_prefetch(&A[(size_t)(m0 + row) * K + k0 + BK], 0, 3);
      __builtin_prefetch(&B[(size_t)(n0 + row) * K + k0 + BK], 0, 3);
    }
    copy_fence();
    __syncthreads();

    V16 af[4], bf[2];
    #pragma unroll
    for (int i = 0; i < 4; ++i) {                     // A frags: lane holds one row
      int r = wm * 64 + i * 16 + lrow;
      af[i].h[0] = *(const v8h*)&As[r * LDT + half];
      af[i].h[1] = *(const v8h*)&As[r * LDT + 16 + half];
    }
    #pragma unroll
    for (int j = 0; j < 2; ++j) {                     // B frags: lane holds one column
      int r = wn * 32 + j * 16 + lrow;
      bf[j].h[0] = *(const v8h*)&Bs[r * LDT + half];
      bf[j].h[1] = *(const v8h*)&Bs[r * LDT + 16 + half];
    }
    #pragma unroll
    for (int i = 0; i < 4; ++i)
      #pragma unroll
      for (int j = 0; j < 2; ++j)
        acc[i][j] = wmma16(af[i].v, bf[j].v, acc[i][j]);
  }

  // epilogue: C layout lanes0-15 -> M = mb+r, lanes16-31 -> M = mb+8+r; N = nb+lrow
  #pragma unroll
  for (int i = 0; i < 4; ++i) {
    #pragma unroll
    for (int j = 0; j < 2; ++j) {
      int n  = n0 + wn * 32 + j * 16 + lrow;
      int mb = m0 + wm * 64 + i * 16 + ((lane < 16) ? 0 : 8);
      float bv = bias[n];
      #pragma unroll
      for (int r = 0; r < 8; ++r) {
        float v = acc[i][j][r] + bv;
        size_t idx = (size_t)(mb + r) * N + n;
        if (EPI == 2) v = 0.5f * v * (1.0f + erff(v * 0.7071067811865476f));
        if (EPI == 1) Cf[idx] = v;
        else          Ch[idx] = (_Float16)v;
      }
    }
  }
}

// ---------------------------------------------------------------- flash attention
// grid (S/64, H, B), 128 threads; each wave owns 16 query rows, streams keys in 32s.
// vT is V transposed: vT[d][m]  (d in [0,DM), m in [0,MROWS)) -> PV B-frags are
// K-contiguous vectorized global loads, no LDS gather needed.
__global__ __launch_bounds__(128) void attn_kernel(
    const _Float16* __restrict__ qh, const _Float16* __restrict__ kh,
    const _Float16* __restrict__ vT, _Float16* __restrict__ ctxh)
{
  const int tid  = threadIdx.x;
  const int lane = tid & 31, wave = tid >> 5;
  const int h = blockIdx.y, b = blockIdx.z;
  const int qbase = b * S_LEN + blockIdx.x * 64 + wave * 16;  // global row
  const int col0  = h * DKH;
  const int lrow  = lane & 15;
  const int half  = (lane < 16) ? 0 : 8;

  __shared__ _Float16 pbuf[4][16][32];   // wave-private P re-swizzle buffer

  V16 qa[2];                             // Q fragments, resident whole kernel
  #pragma unroll
  for (int c = 0; c < 2; ++c) {
    const _Float16* qp = qh + (size_t)(qbase + lrow) * DM + col0 + c * 32;
    qa[c].h[0] = *(const v8h*)(qp + half);
    qa[c].h[1] = *(const v8h*)(qp + 16 + half);
  }

  float mrow[8], lsum[8];
  v8f ctx[4] = {};
  #pragma unroll
  for (int r = 0; r < 8; ++r) { mrow[r] = -3.0e38f; lsum[r] = 0.f; }

  const int kr0 = b * S_LEN;
  const float sc = 0.125f;               // 1/sqrt(64)

  for (int j0 = 0; j0 < S_LEN; j0 += 32) {
    // scores: two 16-key tiles, K-dim = 64 in two 32-chunks
    v8f s[2];
    #pragma unroll
    for (int kb = 0; kb < 2; ++kb) {
      v8f st = {};
      #pragma unroll
      for (int c = 0; c < 2; ++c) {
        const _Float16* kp =
            kh + (size_t)(kr0 + j0 + kb * 16 + lrow) * DM + col0 + c * 32;
        V16 kf;
        kf.h[0] = *(const v8h*)(kp + half);
        kf.h[1] = *(const v8h*)(kp + 16 + half);
        st = wmma16(qa[c].v, kf.v, st);
      }
      s[kb] = st;
    }
    // online softmax (16-lane halves match C-matrix row distribution)
    #pragma unroll
    for (int r = 0; r < 8; ++r) {
      float bm = fmaxf(s[0][r], s[1][r]) * sc;
      #pragma unroll
      for (int m = 1; m < 16; m <<= 1) bm = fmaxf(bm, __shfl_xor(bm, m, 32));
      float mnew  = fmaxf(mrow[r], bm);
      float alpha = expf(mrow[r] - mnew);
      float p0 = expf(s[0][r] * sc - mnew);
      float p1 = expf(s[1][r] * sc - mnew);
      int row = ((lane < 16) ? 0 : 8) + r;
      pbuf[wave][row][lrow]      = (_Float16)p0;
      pbuf[wave][row][16 + lrow] = (_Float16)p1;
      float ps = p0 + p1;
      #pragma unroll
      for (int m = 1; m < 16; m <<= 1) ps += __shfl_xor(ps, m, 32);
      lsum[r] = lsum[r] * alpha + ps;
      mrow[r] = mnew;
      #pragma unroll
      for (int t = 0; t < 4; ++t) ctx[t][r] *= alpha;
    }
    // P fragment (A layout) from LDS
    V16 pf;
    pf.h[0] = *(const v8h*)&pbuf[wave][lrow][half];
    pf.h[1] = *(const v8h*)&pbuf[wave][lrow][16 + half];
    // PV: four 16-wide d_k subtiles; V^T fragments are vectorized global loads
    #pragma unroll
    for (int t = 0; t < 4; ++t) {
      const _Float16* vp =
          vT + (size_t)(col0 + t * 16 + lrow) * MROWS + kr0 + j0;
      V16 vf;
      vf.h[0] = *(const v8h*)(vp + half);
      vf.h[1] = *(const v8h*)(vp + 16 + half);
      ctx[t] = wmma16(pf.v, vf.v, ctx[t]);
    }
  }

  #pragma unroll
  for (int r = 0; r < 8; ++r) {
    float rl = 1.0f / lsum[r];
    int row = qbase + ((lane < 16) ? 0 : 8) + r;
    #pragma unroll
    for (int t = 0; t < 4; ++t)
      ctxh[(size_t)row * DM + col0 + t * 16 + lrow] = (_Float16)(ctx[t][r] * rl);
  }
}

// ---------------------------------------------------------------- residual + LN
__device__ __forceinline__ float block_sum256(float x, float* sm) {
  #pragma unroll
  for (int m = 16; m >= 1; m >>= 1) x += __shfl_xor(x, m, 32);
  int w = threadIdx.x >> 5;
  if ((threadIdx.x & 31) == 0) sm[w] = x;
  __syncthreads();
  float t = 0.f;
  #pragma unroll
  for (int i = 0; i < 8; ++i) t += sm[i];
  __syncthreads();
  return t;
}

__global__ __launch_bounds__(256) void add_ln_kernel(
    const float* __restrict__ a, const float* __restrict__ b,
    const float* __restrict__ g, const float* __restrict__ beta,
    float* __restrict__ out, _Float16* __restrict__ outh)
{
  __shared__ float sm[8];
  const int row = blockIdx.x, tid = threadIdx.x;
  const float* pa = a + (size_t)row * DM;
  const float* pb = b + (size_t)row * DM;
  float v[4]; float s = 0.f;
  #pragma unroll
  for (int i = 0; i < 4; ++i) { int c = tid + i * 256; v[i] = pa[c] + pb[c]; s += v[i]; }
  float mu = block_sum256(s, sm) * (1.0f / DM);
  float q = 0.f;
  #pragma unroll
  for (int i = 0; i < 4; ++i) { float d = v[i] - mu; q += d * d; }
  float var = block_sum256(q, sm) * (1.0f / DM);
  float rs = rsqrtf(var + 1e-5f);
  #pragma unroll
  for (int i = 0; i < 4; ++i) {
    int c = tid + i * 256;
    float o = (v[i] - mu) * rs * g[c] + beta[c];
    out[(size_t)row * DM + c] = o;
    if (outh) outh[(size_t)row * DM + c] = (_Float16)o;
  }
}

// ---------------------------------------------------------------- launch
extern "C" void kernel_launch(void* const* d_in, const int* in_sizes, int n_in,
                              void* d_out, int out_size, void* d_ws, size_t ws_size,
                              hipStream_t stream)
{
  (void)in_sizes; (void)n_in; (void)out_size; (void)ws_size;
  const float* x    = (const float*)d_in[0];
  const float* wq   = (const float*)d_in[1];
  const float* bq   = (const float*)d_in[2];
  const float* wk   = (const float*)d_in[3];
  const float* bk   = (const float*)d_in[4];
  const float* wv   = (const float*)d_in[5];
  const float* bv   = (const float*)d_in[6];
  const float* wo   = (const float*)d_in[7];
  const float* bo   = (const float*)d_in[8];
  const float* ln1g = (const float*)d_in[9];
  const float* ln1b = (const float*)d_in[10];
  const float* w1   = (const float*)d_in[11];
  const float* b1   = (const float*)d_in[12];
  const float* w2   = (const float*)d_in[13];
  const float* b2   = (const float*)d_in[14];
  const float* ln2g = (const float*)d_in[15];
  const float* ln2b = (const float*)d_in[16];

  char* ws = (char*)d_ws;
  const size_t MB = 1024 * 1024;
  _Float16* wqh = (_Float16*)(ws + 0 * MB);
  _Float16* wkh = (_Float16*)(ws + 2 * MB);
  _Float16* wvh = (_Float16*)(ws + 4 * MB);
  _Float16* woh = (_Float16*)(ws + 6 * MB);
  _Float16* w1h = (_Float16*)(ws + 8 * MB);
  _Float16* w2h = (_Float16*)(ws + 16 * MB);
  _Float16* xh  = (_Float16*)(ws + 24 * MB);   // dead after QKV GEMMs
  _Float16* vTh = (_Float16*)(ws + 24 * MB);   // reuses xh slot (after QKV GEMMs)
  _Float16* aoh = (_Float16*)(ws + 24 * MB);   // reuses slot again (after attention)
  _Float16* qh  = (_Float16*)(ws + 32 * MB);
  _Float16* kh  = (_Float16*)(ws + 40 * MB);
  _Float16* vh  = (_Float16*)(ws + 48 * MB);
  _Float16* ch  = (_Float16*)(ws + 56 * MB);
  _Float16* hfh = (_Float16*)(ws + 32 * MB);   // reuses q/k/v/ctx (32MB) after proj
  float*    yf  = (float*)(ws + 64 * MB);      // 16MB f32 GEMM output scratch
  float*    ao  = (float*)(ws + 80 * MB);      // 16MB attn_out f32  -> total 96MB

  auto cvt = [&](const float* s, _Float16* d, int n) {
    cvt_f32_f16<<<(n + 255) / 256, 256, 0, stream>>>(s, d, n);
  };
  cvt(wq, wqh, DM * DM); cvt(wk, wkh, DM * DM);
  cvt(wv, wvh, DM * DM); cvt(wo, woh, DM * DM);
  cvt(w1, w1h, FF * DM); cvt(w2, w2h, DM * FF);
  cvt(x,  xh,  MROWS * DM);

  dim3 g1(DM / 128, MROWS / 128);
  gemm_nt<0><<<g1, 256, 0, stream>>>(xh, wqh, bq, nullptr, qh, MROWS, DM, DM);
  gemm_nt<0><<<g1, 256, 0, stream>>>(xh, wkh, bk, nullptr, kh, MROWS, DM, DM);
  gemm_nt<0><<<g1, 256, 0, stream>>>(xh, wvh, bv, nullptr, vh, MROWS, DM, DM);

  // V^T for the PV matmul (xh slot is dead now)
  transpose_f16<<<dim3(DM / 32, MROWS / 32), 256, 0, stream>>>(vh, vTh, MROWS, DM);

  attn_kernel<<<dim3(S_LEN / 64, HEADS, 2), 128, 0, stream>>>(qh, kh, vTh, ch);

  gemm_nt<1><<<g1, 256, 0, stream>>>(ch, woh, bo, yf, nullptr, MROWS, DM, DM);
  add_ln_kernel<<<MROWS, 256, 0, stream>>>(yf, x, ln1g, ln1b, ao, aoh);

  gemm_nt<2><<<dim3(FF / 128, MROWS / 128), 256, 0, stream>>>(aoh, w1h, b1, nullptr,
                                                              hfh, MROWS, FF, DM);
  gemm_nt<1><<<g1, 256, 0, stream>>>(hfh, w2h, b2, yf, nullptr, MROWS, DM, FF);
  add_ln_kernel<<<MROWS, 256, 0, stream>>>(yf, ao, ln2g, ln2b, (float*)d_out, nullptr);
}
